// Attention_23768349016561
// MI455X (gfx1250) — matrix operand order repaired and checked by reading, not compiled, not run on hardware
//
#include <hip/hip_runtime.h>
#include <hip/hip_bf16.h>

// ---------------------------------------------------------------------------
// CDNA5 (gfx1250) fused multi-head attention, bf16 WMMA path.
//   1. convert x + weights fp32 -> bf16 (workspace)
//   2. Q/K/V projections: bf16 GEMM (v_wmma_f32_16x16x32_bf16), bias add,
//      head-split store (V stored transposed [B,H,Dh,N]); the softmax scale
//      Dh^-0.5 * log2(e) is folded into the Q projection epilogue.
//   3. flash attention computed TRANSPOSED: S^T = K Q^T so that the softmax
//      axis (keys) lies along VGPRs within a lane; P^T feeds O^T = V^T P^T
//      directly as a WMMA B-matrix (no LDS transpose, no barriers).
//   4. output projection: bf16 GEMM -> fp32 d_out
// ---------------------------------------------------------------------------

typedef __attribute__((ext_vector_type(16))) __bf16 v16bf;
typedef __attribute__((ext_vector_type(2)))  __bf16 v2bf;
typedef __attribute__((ext_vector_type(8)))  float  v8f;

union Frag {
  v16bf bf;
  uint4 u4[2];
  unsigned int u32[8];
  unsigned short us[16];
};

// Native bf16 conversion (RNE) -> v_cvt_*bf16* on gfx1250.
__device__ __forceinline__ unsigned int pack_bf16x2(float lo, float hi) {
  union { v2bf v; unsigned int u; } x;
  x.v.x = (__bf16)lo;
  x.v.y = (__bf16)hi;
  return x.u;
}

__device__ __forceinline__ v8f wmma_bf16(const Frag& a, const Frag& b, v8f c) {
  return __builtin_amdgcn_wmma_f32_16x16x32_bf16(false, a.bf, false, b.bf,
                                                 (short)0, c, false, false);
}

__global__ __launch_bounds__(256) void convert_f32_to_bf16(
    const float* __restrict__ src, __bf16* __restrict__ dst, int n) {
  int i = blockIdx.x * blockDim.x + threadIdx.x;
  int stride = gridDim.x * blockDim.x;
  for (; i < n; i += stride) dst[i] = (__bf16)src[i];
}

// ---------------------------------------------------------------------------
// GEMM: Out[m,c] = (sum_k A[m,k] * Bw[c,k] + bias[c]) * oscale
// A: [M,K] bf16 row-major.  Bw: [COLS,K] bf16 row-major (torch Linear W).
// Block = 256 threads (8 waves), block tile 128x128, wave tile 32x64.
// ---------------------------------------------------------------------------
template <bool F32OUT>
__global__ __launch_bounds__(256) void gemm_bf16_wmma(
    const unsigned short* __restrict__ A,
    const unsigned short* __restrict__ Bw,
    const float* __restrict__ bias,
    void* __restrict__ Out,
    int M, int K, int COLS, float oscale,
    long sb, long sh, long sn, long sd) {
  const int lane  = threadIdx.x & 31;
  const int wave  = threadIdx.x >> 5;
  const int laneN = lane & 15;
  const int hi    = lane >> 4;   // lane half: selects k sub-chunk per ISA layout

  const int m0 = blockIdx.x * 128 + (wave & 3) * 32;
  const int c0 = blockIdx.y * 128 + (wave >> 2) * 64;

  v8f acc[2][4];
#pragma unroll
  for (int i = 0; i < 2; ++i)
#pragma unroll
    for (int f = 0; f < 4; ++f) acc[i][f] = {};

  const unsigned short* arow0 = A + (size_t)(m0 + laneN) * K;
  const unsigned short* arow1 = arow0 + (size_t)16 * K;

#pragma unroll 2
  for (int k0 = 0; k0 < K; k0 += 32) {
    // Cover the next K-tile of the streamed activations while WMMAs run.
    __builtin_prefetch(arow0 + k0 + 32, 0, 0);
    __builtin_prefetch(arow1 + k0 + 32, 0, 0);
    // A frags 16x32 bf16: lane=row (mod 16); lo half holds k {0..7,16..23},
    // hi half holds k {8..15,24..31}  -> two b128 loads per frag.
    Frag a0, a1;
    a0.u4[0] = *(const uint4*)(arow0 + k0 + hi * 8);
    a0.u4[1] = *(const uint4*)(arow0 + k0 + 16 + hi * 8);
    a1.u4[0] = *(const uint4*)(arow1 + k0 + hi * 8);
    a1.u4[1] = *(const uint4*)(arow1 + k0 + 16 + hi * 8);
    // B frags 32x16: lane=col, lo half k {0..7,16..23}, hi half k {8..15,24..31}.
    Frag b[4];
#pragma unroll
    for (int f = 0; f < 4; ++f) {
      const unsigned short* bp =
          Bw + (size_t)(c0 + f * 16 + laneN) * K + k0 + hi * 8;
      b[f].u4[0] = *(const uint4*)(bp);
      b[f].u4[1] = *(const uint4*)(bp + 16);
    }
#pragma unroll
    for (int f = 0; f < 4; ++f) {
      acc[0][f] = wmma_bf16(a0, b[f], acc[0][f]);
      acc[1][f] = wmma_bf16(a1, b[f], acc[1][f]);
    }
  }

  // Epilogue. C/D layout: lane = col (mod 16), VGPR r = row, lane-half = row+8.
#pragma unroll
  for (int f = 0; f < 4; ++f) {
    const int c = c0 + f * 16 + laneN;
    const float bvv = bias[c];
#pragma unroll
    for (int i = 0; i < 2; ++i) {
#pragma unroll
      for (int r = 0; r < 8; ++r) {
        const int row = m0 + i * 16 + r + 8 * hi;
        const float val = (acc[i][f][r] + bvv) * oscale;
        if constexpr (F32OUT) {
          ((float*)Out)[(size_t)row * COLS + c] = val;
        } else {
          const int bidx = row >> 11, n = row & 2047;
          const int h = c >> 6, d = c & 63;
          ((__bf16*)Out)[(size_t)bidx * sb + (size_t)h * sh +
                         (size_t)n * sn + (size_t)d * sd] = (__bf16)val;
        }
      }
    }
  }
}

// ---------------------------------------------------------------------------
// Transposed flash attention. Block = 128 threads (4 independent waves);
// each wave owns 16 query rows of one (b,h). Keys streamed 64 at a time.
//   S^T tile: A = K-tile (16 keys x 32 d), B = Q^T (32 d x 16 q)
//     -> C layout: lane = q column, VGPR r = key (r + 8*laneHalf) per tile.
//   Softmax over keys = in-register tree + ONE xor-16 shuffle per pass.
//   Q was pre-scaled by Dh^-0.5*log2(e), so scores are already exp2 exponents.
//   O^T: A = V^T tile (16 d x 32 keys), B = P^T (32 keys x 16 q); the P^T
//   B-fragment needs only an 8-dword xor-16 exchange with the partner half.
// Q,K: [B*H, N, 64] bf16.  Vt: [B*H, 64, N] bf16.  O: [B, N, H*64] bf16.
// ---------------------------------------------------------------------------
#define ATT_N  2048
#define ATT_DH 64

__global__ __launch_bounds__(128) void attn_fwd(
    const unsigned short* __restrict__ Q,
    const unsigned short* __restrict__ Kt,
    const unsigned short* __restrict__ Vt,
    unsigned short* __restrict__ O) {
  const int lane  = threadIdx.x & 31;
  const int wave  = threadIdx.x >> 5;
  const int laneN = lane & 15;
  const int hi    = lane >> 4;

  const int qb = blockIdx.x & 31;   // 32 q-blocks of 64 rows
  const int bh = blockIdx.x >> 5;   // 0..63 = b*16+h
  const int q0 = qb * 64 + wave * 16;

  const unsigned short* qbase = Q  + (size_t)bh * ATT_N * ATT_DH;
  const unsigned short* kbase = Kt + (size_t)bh * ATT_N * ATT_DH;
  const unsigned short* vbase = Vt + (size_t)bh * ATT_DH * ATT_N;

  // Q^T B-frags (lane = q col, d {8h..8h+7, 16+8h..16+8h+7} per half), loaded once.
  Frag qbf[2];
  {
    const unsigned short* qrow = qbase + (size_t)(q0 + laneN) * ATT_DH;
#pragma unroll
    for (int j = 0; j < 2; ++j) {
      qbf[j].u4[0] = *(const uint4*)(qrow + j * 32 + hi * 8);
      qbf[j].u4[1] = *(const uint4*)(qrow + j * 32 + hi * 8 + 16);
    }
  }

  v8f acc[4];   // O^T tiles: lane = q col, VGPR r -> d = dt*16 + r + 8*hi
#pragma unroll
  for (int dt = 0; dt < 4; ++dt) acc[dt] = {};
  float mrun = -1.0e30f, lrun = 0.0f;  // per-lane scalars (one q column)

  for (int kb = 0; kb < ATT_N; kb += 64) {
    // --- S^T tiles: s[t] holds keys kb + t*16 + r + 8*hi for q = laneN ---
    v8f s[4];
#pragma unroll
    for (int t = 0; t < 4; ++t) {
      const unsigned short* kr =
          kbase + (size_t)(kb + t * 16 + laneN) * ATT_DH;
      Frag ka0, ka1;  // A frags of K-tile: d chunks 0..31, 32..63
      ka0.u4[0] = *(const uint4*)(kr + hi * 8);
      ka0.u4[1] = *(const uint4*)(kr + 16 + hi * 8);
      ka1.u4[0] = *(const uint4*)(kr + 32 + hi * 8);
      ka1.u4[1] = *(const uint4*)(kr + 32 + 16 + hi * 8);
      v8f st = {};
      st = wmma_bf16(ka0, qbf[0], st);
      st = wmma_bf16(ka1, qbf[1], st);
      s[t] = st;
    }

    // --- softmax over keys: in-register tree + one xor-16 shuffle ---
    float rmax = -1.0e30f;
#pragma unroll
    for (int t = 0; t < 4; ++t)
#pragma unroll
      for (int r = 0; r < 8; ++r) rmax = fmaxf(rmax, s[t][r]);
    rmax = fmaxf(rmax, __shfl_xor(rmax, 16, 32));
    const float mnew = fmaxf(mrun, rmax);
    const float fsc = exp2f(mrun - mnew);
    mrun = mnew;

    // exp + pack P^T as bf16 pairs; accumulate local row sum
    unsigned int pk[4][4];
    float psum = 0.0f;
#pragma unroll
    for (int t = 0; t < 4; ++t)
#pragma unroll
      for (int i = 0; i < 4; ++i) {
        const float e0 = exp2f(s[t][2 * i]     - mnew);
        const float e1 = exp2f(s[t][2 * i + 1] - mnew);
        psum += e0 + e1;
        pk[t][i] = pack_bf16x2(e0, e1);
      }
    psum += __shfl_xor(psum, 16, 32);
    lrun = lrun * fsc + psum;
#pragma unroll
    for (int dt = 0; dt < 4; ++dt)
#pragma unroll
      for (int r = 0; r < 8; ++r) acc[dt][r] = acc[dt][r] * fsc;

    // --- build P^T B-frags: partner half-lane holds the other 8 keys ---
    Frag pb0, pb1;
#pragma unroll
    for (int i = 0; i < 4; ++i) {
      const unsigned int s0 = hi ? pk[0][i] : pk[1][i];
      const unsigned int s1 = hi ? pk[2][i] : pk[3][i];
      const unsigned int r0 = __shfl_xor(s0, 16, 32);
      const unsigned int r1 = __shfl_xor(s1, 16, 32);
      pb0.u32[i]     = hi ? r0 : pk[0][i];
      pb0.u32[4 + i] = hi ? pk[1][i] : r0;
      pb1.u32[i]     = hi ? r1 : pk[2][i];
      pb1.u32[4 + i] = hi ? pk[3][i] : r1;
    }

    // --- O^T += V^T P^T : A = V^T tiles (contiguous keys per d row) ---
#pragma unroll
    for (int dt = 0; dt < 4; ++dt) {
      const unsigned short* vr =
          vbase + (size_t)(dt * 16 + laneN) * ATT_N + kb;
      Frag va0, va1;
      va0.u4[0] = *(const uint4*)(vr + hi * 16);
      va0.u4[1] = *(const uint4*)(vr + hi * 16 + 8);
      va1.u4[0] = *(const uint4*)(vr + 32 + hi * 16);
      va1.u4[1] = *(const uint4*)(vr + 32 + hi * 16 + 8);
      acc[dt] = wmma_bf16(va0, pb0, acc[dt]);
      acc[dt] = wmma_bf16(va1, pb1, acc[dt]);
    }
  }

  // --- O = acc / l; consecutive VGPRs are consecutive d -> b128 stores ---
  const int b = bh >> 4, h = bh & 15;
  const float linv = 1.0f / lrun;
  unsigned short* orow =
      O + ((size_t)b * ATT_N + (q0 + laneN)) * 1024 + h * 64;
#pragma unroll
  for (int dt = 0; dt < 4; ++dt) {
    Frag ov;
#pragma unroll
    for (int i = 0; i < 4; ++i)
      ov.u32[i] = pack_bf16x2(acc[dt][2 * i] * linv, acc[dt][2 * i + 1] * linv);
    *(uint4*)(orow + dt * 16 + 8 * hi) = ov.u4[0];
  }
}

// ---------------------------------------------------------------------------
extern "C" void kernel_launch(void* const* d_in, const int* in_sizes, int n_in,
                              void* d_out, int out_size, void* d_ws,
                              size_t ws_size, hipStream_t stream) {
  (void)in_sizes; (void)n_in; (void)out_size; (void)ws_size;
  const float* x  = (const float*)d_in[0];
  const float* Wq = (const float*)d_in[1];
  const float* bq = (const float*)d_in[2];
  const float* Wk = (const float*)d_in[3];
  const float* bk = (const float*)d_in[4];
  const float* Wv = (const float*)d_in[5];
  const float* bv = (const float*)d_in[6];
  const float* Wo = (const float*)d_in[7];
  const float* bo = (const float*)d_in[8];

  const int B = 4, N = 2048, C = 1024, H = 16, Dh = 64;
  const size_t MR = (size_t)B * N;  // 8192 rows

  char* p = (char*)d_ws;
  unsigned short* xbf = (unsigned short*)p; p += MR * C * 2;           // 16 MB
  unsigned short* wqb = (unsigned short*)p; p += (size_t)C * C * 2;    //  2 MB
  unsigned short* wkb = (unsigned short*)p; p += (size_t)C * C * 2;
  unsigned short* wvb = (unsigned short*)p; p += (size_t)C * C * 2;
  unsigned short* wob = (unsigned short*)p; p += (size_t)C * C * 2;
  unsigned short* qws = (unsigned short*)p; p += MR * C * 2;           // 16 MB
  unsigned short* kws = (unsigned short*)p; p += MR * C * 2;
  unsigned short* vws = (unsigned short*)p; p += MR * C * 2;
  unsigned short* ows = (unsigned short*)p; p += MR * C * 2;

  convert_f32_to_bf16<<<1024, 256, 0, stream>>>(x, (__bf16*)xbf, (int)(MR * C));
  convert_f32_to_bf16<<<256, 256, 0, stream>>>(Wq, (__bf16*)wqb, C * C);
  convert_f32_to_bf16<<<256, 256, 0, stream>>>(Wk, (__bf16*)wkb, C * C);
  convert_f32_to_bf16<<<256, 256, 0, stream>>>(Wv, (__bf16*)wvb, C * C);
  convert_f32_to_bf16<<<256, 256, 0, stream>>>(Wo, (__bf16*)wob, C * C);

  const dim3 g(64, 8);  // 8192/128 x 1024/128
  const long sb = (long)H * N * Dh;  // 2097152
  const long sh = (long)N * Dh;      //  131072
  // Softmax scale folded into Q: Dh^-0.5 * log2(e)
  const float qscale = 0.125f * 1.44269504088896340736f;
  // Q, K: [B,H,N,Dh]
  gemm_bf16_wmma<false><<<g, 256, 0, stream>>>(xbf, wqb, bq, qws, 8192, 1024,
                                               1024, qscale, sb, sh, (long)Dh,
                                               1L);
  gemm_bf16_wmma<false><<<g, 256, 0, stream>>>(xbf, wkb, bk, kws, 8192, 1024,
                                               1024, 1.0f, sb, sh, (long)Dh,
                                               1L);
  // V transposed: [B,H,Dh,N]
  gemm_bf16_wmma<false><<<g, 256, 0, stream>>>(xbf, wvb, bv, vws, 8192, 1024,
                                               1024, 1.0f, sb, sh, 1L, (long)N);

  attn_fwd<<<B * H * (N / 64), 128, 0, stream>>>(qws, kws, vws, ows);

  gemm_bf16_wmma<true><<<g, 256, 0, stream>>>(ows, wob, bo, d_out, 8192, 1024,
                                              1024, 1.0f, 0, 0, 0, 0);
}
